// SelfAttention_7404523619225
// MI455X (gfx1250) — matrix operand order repaired and checked by reading, not compiled, not run on hardware
//
#include <hip/hip_runtime.h>
#include <cmath>

#define USE_TDM 1

typedef __bf16 bf16_t;
typedef __attribute__((ext_vector_type(16))) __bf16 v16bf;
typedef __attribute__((ext_vector_type(8)))  __bf16 v8bf;
typedef __attribute__((ext_vector_type(8)))  float  v8f;
typedef __attribute__((ext_vector_type(4)))  unsigned int u32x4;
typedef __attribute__((ext_vector_type(8)))  unsigned int u32x8;
typedef int b128v __attribute__((vector_size(16)));   // matches builtin param type

static __device__ __forceinline__ v16bf cat8(v8bf lo, v8bf hi) {
  return __builtin_shufflevector(lo, hi, 0,1,2,3,4,5,6,7,8,9,10,11,12,13,14,15);
}
static __device__ __forceinline__ v8f wmma_bf16(v16bf a, v16bf b, v8f c) {
  // (neg_a, A, neg_b, B, c_mod, C, reuse_a, reuse_b)
  return __builtin_amdgcn_wmma_f32_16x16x32_bf16(false, a, false, b, (short)0, c, false, false);
}

// ---- async global->LDS copy (ASYNCcnt path), guarded fallback to sync copy
#if __has_builtin(__builtin_amdgcn_global_load_async_to_lds_b128)
#define HAVE_ASYNC 1
static __device__ __forceinline__ void cp16_g2l(const bf16_t* g, bf16_t* l) {
  __builtin_amdgcn_global_load_async_to_lds_b128(
      (__attribute__((address_space(1))) b128v*)g,
      (__attribute__((address_space(3))) b128v*)l, 0, 0);
}
#else
#define HAVE_ASYNC 0
static __device__ __forceinline__ void cp16_g2l(const bf16_t* g, bf16_t* l) {
  *(v8bf*)l = *(const v8bf*)g;
}
#endif

static __device__ __forceinline__ void wait_async0() {
#if HAVE_ASYNC
#if __has_builtin(__builtin_amdgcn_s_wait_asynccnt)
  __builtin_amdgcn_s_wait_asynccnt(0);
#else
  asm volatile("s_wait_asynccnt 0x0" ::: "memory");
#endif
#endif
}

static __device__ __forceinline__ void wait_tensor0() {
#if __has_builtin(__builtin_amdgcn_s_wait_tensorcnt)
  __builtin_amdgcn_s_wait_tensorcnt(0);
#else
  asm volatile("s_wait_tensorcnt 0x0" ::: "memory");
#endif
}

// ---------------------------------------------------------------- convert
__global__ void cvt_f32_bf16(const float* __restrict__ in, bf16_t* __restrict__ out, int n) {
  int i = blockIdx.x * blockDim.x + threadIdx.x;
  if (i < n) out[i] = (bf16_t)in[i];
}

// ---------------------------------------------------------------- GEMM: C[M,N] = scale*(A[M,K] @ B[N,K]^T) (+bias)
// A, B row-major bf16 with K contiguous. Block tile 128x64, 8 waves (4x2), each wave 32x32.
#define GTM 128
#define GTN 64
#define GTK 32

template<bool OUTF32, bool BIAS>
__global__ void gemm_nt_bf16(const bf16_t* __restrict__ A, const bf16_t* __restrict__ B,
                             void* __restrict__ Cv, const float* __restrict__ bias,
                             int M, int N, int K, float scale)
{
  __shared__ __align__(16) bf16_t As[GTM * GTK];
  __shared__ __align__(16) bf16_t Bs[GTN * GTK];

  const int t    = threadIdx.x;
  const int lane = t & 31;
  const int w    = t >> 5;            // 0..7
  const int mw   = (w >> 1) * 32;     // 0,32,64,96
  const int nw   = (w & 1) * 32;      // 0,32
  const int l15  = lane & 15;
  const int lhi  = lane >> 4;         // 0 or 1
  const int mbase = blockIdx.y * GTM;
  const int nbase = blockIdx.x * GTN;

  v8f c[2][2];
  for (int i = 0; i < 2; i++)
    for (int j = 0; j < 2; j++)
      for (int r = 0; r < 8; r++) c[i][j][r] = 0.0f;

  for (int k0 = 0; k0 < K; k0 += GTK) {
    // stage A tile 128x32 (512 slots of 8 bf16) via async LDS DMA
    for (int s = t; s < (GTM * GTK) / 8; s += 256) {
      int row = s >> 2, ch = s & 3;
      cp16_g2l(A + (size_t)(mbase + row) * K + k0 + ch * 8, &As[row * GTK + ch * 8]);
    }
    // stage B tile 64x32 (256 slots)
    for (int s = t; s < (GTN * GTK) / 8; s += 256) {
      int row = s >> 2, ch = s & 3;
      cp16_g2l(B + (size_t)(nbase + row) * K + k0 + ch * 8, &Bs[row * GTK + ch * 8]);
    }
    wait_async0();
    __syncthreads();

    v16bf af[2], bf[2];
#pragma unroll
    for (int i = 0; i < 2; i++) {
      const bf16_t* ar = &As[(mw + i * 16 + l15) * GTK];
      af[i] = cat8(*(const v8bf*)(ar + lhi * 16), *(const v8bf*)(ar + lhi * 16 + 8));
    }
#pragma unroll
    for (int j = 0; j < 2; j++) {
      const bf16_t* br = &Bs[(nw + j * 16 + l15) * GTK];
      bf[j] = cat8(*(const v8bf*)(br + lhi * 16), *(const v8bf*)(br + lhi * 16 + 8));
    }
#pragma unroll
    for (int i = 0; i < 2; i++)
#pragma unroll
      for (int j = 0; j < 2; j++) c[i][j] = wmma_bf16(af[i], bf[j], c[i][j]);
    __syncthreads();
  }

  // epilogue: C layout — lane holds column (l15), 8 rows (r + lhi*8) per frag
  for (int i = 0; i < 2; i++)
    for (int j = 0; j < 2; j++) {
      int n  = nbase + nw + j * 16 + l15;
      int m0 = mbase + mw + i * 16 + (lhi << 3);
      float bv = BIAS ? bias[n] : 0.0f;
#pragma unroll
      for (int r = 0; r < 8; r++) {
        float v = c[i][j][r] * scale + bv;
        if (OUTF32) ((float*)Cv)[(size_t)(m0 + r) * N + n] = v;
        else        ((bf16_t*)Cv)[(size_t)(m0 + r) * N + n] = (bf16_t)v;
      }
    }
}

// ---------------------------------------------------------------- streaming attention
// Grid: (S/32 query tiles, B*H). Block 256 thr / 8 waves.
// Per iter: 64-key tile. Waves compute 8 score tiles (qg x kg = 2x4), online
// softmax in LDS, then 8 waves update O (qg x cg = 2x4, 128 cols each).
#define AS_ 2048
#define AD_ 512
#define AEH 4096

__global__ void attn_flash_bf16(const bf16_t* __restrict__ Q, const bf16_t* __restrict__ Kb,
                                const bf16_t* __restrict__ V, bf16_t* __restrict__ O)
{
  extern __shared__ __align__(16) char smem[];
  bf16_t* Qs  = (bf16_t*)smem;                       // 32*512  bf16 = 32768 B
  bf16_t* Vt  = (bf16_t*)(smem + 32768);             // 512*64  bf16 = 65536 B (transposed V tile)
  float*  Sc  = (float*)(smem + 32768 + 65536);      // 32*64   f32  = 8192 B
  bf16_t* P   = (bf16_t*)(smem + 32768 + 65536 + 8192); // 32*64 bf16 = 4096 B
  float*  mrow = (float*)(smem + 32768 + 65536 + 8192 + 4096);
  float*  lrow = mrow + 32;
  float*  arow = lrow + 32;

  const int t    = threadIdx.x;
  const int lane = t & 31;
  const int w    = t >> 5;
  const int l15  = lane & 15;
  const int lhi  = lane >> 4;
  const int qg   = w >> 2;   // 0..1 (16-row query group)
  const int kg   = w & 3;    // score phase: 16-key group
  const int cg   = w & 3;    // O phase: 128-col group

  const int qtile = blockIdx.x * 32;
  const int bh = blockIdx.y;
  const int b = bh >> 3, h = bh & 7;
  const size_t base = ((size_t)b * AS_) * AEH + (size_t)h * AD_;
  const bf16_t* Qh = Q + base;
  const bf16_t* Kh = Kb + base;
  const bf16_t* Vh = V + base;

#if USE_TDM
  // ---- stage Q tile (32 x 512, row stride 4096) via Tensor Data Mover
  if (w == 0) {
    unsigned lds_off = (unsigned)(unsigned long long)
        (__attribute__((address_space(3))) char*)(void*)Qs;
    unsigned long long ga = (unsigned long long)(const void*)(Qh + (size_t)qtile * AEH);
    u32x4 g0;
    g0[0] = 1u;                                      // count=1, user descriptor
    g0[1] = lds_off;                                 // lds_addr
    g0[2] = (unsigned)(ga & 0xffffffffu);            // global_addr[31:0]
    g0[3] = (unsigned)((ga >> 32) & 0x01ffffffu) | (2u << 30); // global_addr[56:32] | type=2
    u32x8 g1;
    g1[0] = (1u << 16);                              // wg_mask=0, data_size=1 (2 bytes)
    g1[1] = (unsigned)((AD_ & 0xffffu) << 16);       // tensor_dim0 lo16 (=512)
    g1[2] = (unsigned)((AD_ >> 16) | ((AS_ & 0xffffu) << 16)); // td0 hi | tensor_dim1 lo (=2048)
    g1[3] = (unsigned)((AS_ >> 16) | (AD_ << 16));   // td1 hi | tile_dim0 = 512
    g1[4] = 32u;                                     // tile_dim1 = 32, tile_dim2 = 0
    g1[5] = (unsigned)AEH;                           // tensor_dim0_stride lo32 (=4096)
    g1[6] = 0u;                                      // td0_stride hi | td1_stride lo
    g1[7] = 0u;                                      // td1_stride hi
    asm volatile("tensor_load_to_lds %0, %1" :: "s"(g0), "s"(g1) : "memory");
    wait_tensor0();
  }
#else
  for (int s = t; s < (32 * AD_) / 8; s += 256) {
    int row = s >> 6, ch = s & 63;
    cp16_g2l(Qh + (size_t)(qtile + row) * AEH + ch * 8, &Qs[row * AD_ + ch * 8]);
  }
  wait_async0();
#endif
  if (t < 32) { mrow[t] = -3.0e38f; lrow[t] = 0.0f; }

  v8f oc[8];
  for (int j = 0; j < 8; j++)
    for (int r = 0; r < 8; r++) oc[j][r] = 0.0f;

  __syncthreads();

  for (int kb = 0; kb < AS_; kb += 64) {
    // ---- stage V tile transposed: Vt[e][k], e in [0,512), k in [0,64)
    for (int i = 0; i < 16; i++) {
      int s  = i * 256 + t;      // 4096 slots of 8
      int k  = s >> 6;
      int ch = s & 63;
      v8bf vv = *(const v8bf*)(Vh + (size_t)(kb + k) * AEH + ch * 8);
#pragma unroll
      for (int j = 0; j < 8; j++) Vt[(ch * 8 + j) * 64 + k] = vv[j];
    }

    // ---- scores: wave (qg,kg) computes its 16x16 tile over e=512
    {
      v8f sfr;
      for (int r = 0; r < 8; r++) sfr[r] = 0.0f;
      const int qrow = qg * 16 + l15;
      const int key  = kb + kg * 16 + l15;
      const bf16_t* Kr = Kh + (size_t)key * AEH;
#pragma unroll 4
      for (int ec = 0; ec < AD_; ec += 32) {
        const bf16_t* ar = &Qs[qrow * AD_ + ec];
        v16bf af = cat8(*(const v8bf*)(ar + lhi * 16), *(const v8bf*)(ar + lhi * 16 + 8));
        const bf16_t* br = Kr + ec + lhi * 16;
        v16bf bfv = cat8(*(const v8bf*)br, *(const v8bf*)(br + 8));
        sfr = wmma_bf16(af, bfv, sfr);
      }
      int col = kg * 16 + l15;
      int r0  = qg * 16 + (lhi << 3);
#pragma unroll
      for (int r = 0; r < 8; r++) Sc[(r0 + r) * 64 + col] = sfr[r];
    }
    __syncthreads();

    // ---- online softmax, one thread per query row
    if (t < 32) {
      float mo = mrow[t];
      float mx = mo;
      for (int j = 0; j < 64; j++) mx = fmaxf(mx, Sc[t * 64 + j]);
      float al  = __expf(mo - mx);
      float sum = 0.0f;
      for (int j = 0; j < 64; j++) {
        float p = __expf(Sc[t * 64 + j] - mx);
        P[t * 64 + j] = (bf16_t)p;
        sum += p;
      }
      lrow[t] = lrow[t] * al + sum;
      mrow[t] = mx;
      arow[t] = al;
    }
    __syncthreads();

    // ---- O update: wave (qg,cg) owns 16 x 128 slice of O accumulators
    {
#pragma unroll
      for (int r = 0; r < 8; r++) {
        float al = arow[qg * 16 + (lhi << 3) + r];
#pragma unroll
        for (int j = 0; j < 8; j++) oc[j][r] *= al;
      }
#pragma unroll
      for (int kc = 0; kc < 64; kc += 32) {
        const bf16_t* pr = &P[(qg * 16 + l15) * 64 + kc];
        v16bf af = cat8(*(const v8bf*)(pr + lhi * 16), *(const v8bf*)(pr + lhi * 16 + 8));
        // load all 8 B fragments first so WMMAs can issue back-to-back
        v16bf bfv[8];
#pragma unroll
        for (int j = 0; j < 8; j++) {
          const bf16_t* br = &Vt[(cg * 128 + j * 16 + l15) * 64 + kc + lhi * 16];
          bfv[j] = cat8(*(const v8bf*)br, *(const v8bf*)(br + 8));
        }
#pragma unroll
        for (int j = 0; j < 8; j++) oc[j] = wmma_bf16(af, bfv[j], oc[j]);
      }
    }
    __syncthreads();
  }

  // ---- finalize: O[q][h*512+e] = oc / l
  {
    bf16_t* Oh = O + base;
    int r0 = qg * 16 + (lhi << 3);
#pragma unroll
    for (int r = 0; r < 8; r++) {
      float linv = 1.0f / lrow[r0 + r];
      int q = qtile + r0 + r;
#pragma unroll
      for (int j = 0; j < 8; j++) {
        int e = cg * 128 + j * 16 + l15;
        Oh[(size_t)q * AEH + e] = (bf16_t)(oc[j][r] * linv);
      }
    }
  }
}

// ---------------------------------------------------------------- launch
extern "C" void kernel_launch(void* const* d_in, const int* in_sizes, int n_in,
                              void* d_out, int out_size, void* d_ws, size_t ws_size,
                              hipStream_t stream) {
  const float* x  = (const float*)d_in[0];
  const float* Wq = (const float*)d_in[1];
  const float* Wk = (const float*)d_in[2];
  const float* Wv = (const float*)d_in[3];
  const float* Wu = (const float*)d_in[4];
  const float* bu = (const float*)d_in[5];

  const int Bn = 8, S = 2048, E = 512, H = 8;
  const int M  = Bn * S;   // 16384
  const int EH = E * H;    // 4096

  char* ws = (char*)d_ws;
  size_t off = 0;
  auto alloc = [&](size_t bytes) {
    char* p = ws + off;
    off += (bytes + 255) & ~(size_t)255;
    return p;
  };
  bf16_t* xb  = (bf16_t*)alloc((size_t)M * E * 2);
  bf16_t* wqb = (bf16_t*)alloc((size_t)EH * E * 2);
  bf16_t* wkb = (bf16_t*)alloc((size_t)EH * E * 2);
  bf16_t* wvb = (bf16_t*)alloc((size_t)EH * E * 2);
  bf16_t* wub = (bf16_t*)alloc((size_t)E * EH * 2);
  bf16_t* Qm  = (bf16_t*)alloc((size_t)M * EH * 2);
  bf16_t* Km  = (bf16_t*)alloc((size_t)M * EH * 2);
  bf16_t* Vm  = (bf16_t*)alloc((size_t)M * EH * 2);
  bf16_t* Om  = (bf16_t*)alloc((size_t)M * EH * 2);

  dim3 blk(256);
  // converts
  cvt_f32_bf16<<<dim3((M * E + 255) / 256), blk, 0, stream>>>(x, xb, M * E);
  cvt_f32_bf16<<<dim3((EH * E + 255) / 256), blk, 0, stream>>>(Wq, wqb, EH * E);
  cvt_f32_bf16<<<dim3((EH * E + 255) / 256), blk, 0, stream>>>(Wk, wkb, EH * E);
  cvt_f32_bf16<<<dim3((EH * E + 255) / 256), blk, 0, stream>>>(Wv, wvb, EH * E);
  cvt_f32_bf16<<<dim3((E * EH + 255) / 256), blk, 0, stream>>>(Wu, wub, E * EH);

  const float sc = powf((float)E, -0.25f);
  // projections: [M,E] x [EH,E]^T -> [M,EH]
  gemm_nt_bf16<false, false><<<dim3(EH / GTN, M / GTM), blk, 0, stream>>>(
      xb, wqb, Qm, nullptr, M, EH, E, sc);
  gemm_nt_bf16<false, false><<<dim3(EH / GTN, M / GTM), blk, 0, stream>>>(
      xb, wkb, Km, nullptr, M, EH, E, sc);
  gemm_nt_bf16<false, false><<<dim3(EH / GTN, M / GTM), blk, 0, stream>>>(
      xb, wvb, Vm, nullptr, M, EH, E, 1.0f);

  // streaming attention
  size_t shmem = 32768 + 65536 + 8192 + 4096 + 3 * 32 * 4; // ~110.9 KB
  attn_flash_bf16<<<dim3(S / 32, Bn * H), blk, shmem, stream>>>(Qm, Km, Vm, Om);

  // output projection: [M,EH] x [E,EH]^T + bu -> [M,E] f32
  gemm_nt_bf16<true, true><<<dim3(E / GTN, M / GTM), blk, 0, stream>>>(
      Om, wub, d_out, bu, M, E, EH, 1.0f);
}
